// MultiHeadCrossAttention_55259049230831
// MI455X (gfx1250) — matrix operand order repaired and checked by reading, not compiled, not run on hardware
//
#include <hip/hip_runtime.h>

// ---------------------------------------------------------------------------
// MultiHeadCrossAttention for MI455X (gfx1250), wave32, WMMA bf16 path,
// with CDNA5 async global->LDS staging (ASYNCcnt) for pure-copy tiles.
// ---------------------------------------------------------------------------

typedef __attribute__((ext_vector_type(16))) __bf16 v16bf;
typedef __attribute__((ext_vector_type(8)))  __bf16 v8bf;
typedef __attribute__((ext_vector_type(8)))  float  v8f;

__device__ __forceinline__ unsigned short f2bf(float f) {
  unsigned u = __float_as_uint(f);
  u += 0x7fffu + ((u >> 16) & 1u);           // round-to-nearest-even
  return (unsigned short)(u >> 16);
}
__device__ __forceinline__ float bf2f(unsigned short h) {
  return __uint_as_float(((unsigned)h) << 16);
}
__device__ __forceinline__ unsigned scale2bf(unsigned w, float s) {
  unsigned short lo = f2bf(bf2f((unsigned short)(w & 0xffffu)) * s);
  unsigned short hi = f2bf(bf2f((unsigned short)(w >> 16)) * s);
  return (unsigned)lo | ((unsigned)hi << 16);
}
__device__ __forceinline__ void store8_scaled(unsigned short* d, uint4 v, float s) {
  *reinterpret_cast<unsigned*>(d + 0) = scale2bf(v.x, s);
  *reinterpret_cast<unsigned*>(d + 2) = scale2bf(v.y, s);
  *reinterpret_cast<unsigned*>(d + 4) = scale2bf(v.z, s);
  *reinterpret_cast<unsigned*>(d + 6) = scale2bf(v.w, s);
}
__device__ __forceinline__ v8f vzero() {
  v8f z;
#pragma unroll
  for (int i = 0; i < 8; ++i) z[i] = 0.0f;
  return z;
}

// --- CDNA5 async global->LDS copy (16B per lane), tracked by ASYNCcnt. ------
// LDS byte address is the low 32 bits of the generic shared-space pointer
// (flat addressing).
__device__ __forceinline__ void async_ld_b128(void* lds_dst, const void* gsrc) {
  unsigned lds = (unsigned)(unsigned long long)lds_dst;
  unsigned long long g = (unsigned long long)gsrc;
  asm volatile("global_load_async_to_lds_b128 %0, %1, off"
               :: "v"(lds), "v"(g) : "memory");
}
__device__ __forceinline__ void wait_async0() {
  asm volatile("s_wait_asynccnt 0" ::: "memory");
}

// A-operand fragment (16xK, 16-bit): lanes 0-15 hold K {0..7,16..23}+k0,
// lanes 16-31 hold K {8..15,24..31}+k0 (ISA 7.12.2).
__device__ __forceinline__ v16bf frag_a(const unsigned short* rowp, int k0, int lane) {
  const int kh = (lane < 16) ? 0 : 8;
  v8bf lo = *reinterpret_cast<const v8bf*>(rowp + k0 + kh);
  v8bf hi = *reinterpret_cast<const v8bf*>(rowp + k0 + kh + 16);
  return __builtin_shufflevector(lo, hi, 0,1,2,3,4,5,6,7,8,9,10,11,12,13,14,15);
}
// B-operand fragment (Kx16, 16-bit): lanes 0-15 hold K {0..7,16..23}+k0,
// lanes 16-31 hold K {8..15,24..31}+k0.
__device__ __forceinline__ v16bf frag_b(const unsigned short* rowp, int k0, int lane) {
  const int kb = (lane < 16) ? 0 : 8;
  v8bf lo = *reinterpret_cast<const v8bf*>(rowp + k0 + kb);
  v8bf hi = *reinterpret_cast<const v8bf*>(rowp + k0 + kb + 16);
  return __builtin_shufflevector(lo, hi, 0,1,2,3,4,5,6,7,8,9,10,11,12,13,14,15);
}
__device__ __forceinline__ v8f wmma_bf16(v16bf a, v16bf b, v8f c) {
  return __builtin_amdgcn_wmma_f32_16x16x32_bf16(false, a, false, b, (short)0, c,
                                                 false, false);
}

// ---------------------------------------------------------------------------
// Generic GEMM: C[M,N] = A[M,K] @ W[N,K]^T + bias (+ residual).
// A_BF16: A is bf16 (ushort) staged via async global->LDS; else f32 converted
// while staging. OUT_F32_RES: add residual R (f32) and write f32; else bf16.
// Block 256 thr (8 waves), tile 128x128x32, 8 WMMAs/wave/K-step.
// ---------------------------------------------------------------------------
template <bool A_BF16, bool OUT_F32_RES>
__global__ __launch_bounds__(256) void gemm_wmma_kernel(
    const void* __restrict__ Ap, const float* __restrict__ W,
    const float* __restrict__ bias, const float* __restrict__ R,
    void* __restrict__ Cp, int M, int N, int K) {
  __shared__ unsigned short As[128][40];   // 32 cols + 8 pad (80B rows)
  __shared__ unsigned short Ws[128][40];
  const int tid = threadIdx.x;
  const int lane = tid & 31;
  const int wid = tid >> 5;
  const int wm = wid & 1;          // 2 wave rows  (64 M each)
  const int wn = wid >> 1;         // 4 wave cols  (32 N each)
  const int lrow = lane & 15;
  const int rbase = 8 * (lane >> 4);
  const int m0 = blockIdx.y * 128;
  const int n0 = blockIdx.x * 128;

  v8f acc[4][2];
#pragma unroll
  for (int i = 0; i < 4; ++i)
#pragma unroll
    for (int j = 0; j < 2; ++j) acc[i][j] = vzero();

  for (int k0 = 0; k0 < K; k0 += 32) {
    if (k0 + 32 < K) {  // prefetch next W k-tile (global_prefetch_b8)
      const int tr = tid >> 3;
      __builtin_prefetch(W + (size_t)(n0 + tr) * K + k0 + 32, 0, 3);
      __builtin_prefetch(W + (size_t)(n0 + 64 + tr) * K + k0 + 32, 0, 3);
    }
    {  // stage W tile (f32 -> bf16 through VALU)
      const int tr = tid >> 3;
      const int tc = (tid & 7) << 2;
#pragma unroll
      for (int p = 0; p < 4; ++p) {
        const int row = p * 32 + tr;
        float4 v = *reinterpret_cast<const float4*>(W + (size_t)(n0 + row) * K + k0 + tc);
        uint2 pk;
        pk.x = (unsigned)f2bf(v.x) | ((unsigned)f2bf(v.y) << 16);
        pk.y = (unsigned)f2bf(v.z) | ((unsigned)f2bf(v.w) << 16);
        *reinterpret_cast<uint2*>(&Ws[row][tc]) = pk;
      }
    }
    if constexpr (A_BF16) {
      const unsigned short* Ab = (const unsigned short*)Ap;
      const int tr = tid >> 2;
      const int tc = (tid & 3) << 3;
#pragma unroll
      for (int p = 0; p < 2; ++p) {
        const int row = p * 64 + tr;
        async_ld_b128(&As[row][tc], Ab + (size_t)(m0 + row) * K + k0 + tc);
      }
      wait_async0();
    } else {
      const float* Af = (const float*)Ap;
      const int tr = tid >> 3;
      const int tc = (tid & 7) << 2;
#pragma unroll
      for (int p = 0; p < 4; ++p) {
        const int row = p * 32 + tr;
        float4 v = *reinterpret_cast<const float4*>(Af + (size_t)(m0 + row) * K + k0 + tc);
        uint2 pk;
        pk.x = (unsigned)f2bf(v.x) | ((unsigned)f2bf(v.y) << 16);
        pk.y = (unsigned)f2bf(v.z) | ((unsigned)f2bf(v.w) << 16);
        *reinterpret_cast<uint2*>(&As[row][tc]) = pk;
      }
    }
    __syncthreads();

    v16bf af[4];
#pragma unroll
    for (int i = 0; i < 4; ++i)
      af[i] = frag_a(&As[wm * 64 + i * 16 + lrow][0], 0, lane);
    v16bf bfr[2];
#pragma unroll
    for (int j = 0; j < 2; ++j)
      bfr[j] = frag_b(&Ws[wn * 32 + j * 16 + lrow][0], 0, lane);
#pragma unroll
    for (int i = 0; i < 4; ++i)
#pragma unroll
      for (int j = 0; j < 2; ++j)
        acc[i][j] = wmma_bf16(af[i], bfr[j], acc[i][j]);
    __syncthreads();
  }

#pragma unroll
  for (int j = 0; j < 2; ++j) {
    const int gcol = n0 + wn * 32 + j * 16 + lrow;
    const float bv = bias[gcol];
#pragma unroll
    for (int i = 0; i < 4; ++i) {
#pragma unroll
      for (int r = 0; r < 8; ++r) {
        const int grow = m0 + wm * 64 + i * 16 + rbase + r;
        float val = acc[i][j][r] + bv;
        if constexpr (OUT_F32_RES) {
          val += R[(size_t)grow * N + gcol];
          ((float*)Cp)[(size_t)grow * N + gcol] = val;
        } else {
          ((unsigned short*)Cp)[(size_t)grow * N + gcol] = f2bf(val);
        }
      }
    }
  }
}

// ---------------------------------------------------------------------------
// attn1: flash attention per (b, h, 64-query tile). 128 thr = 4 waves,
// each wave owns 16 q rows. Writes bf16 context + per-row (max, 1/sum) stats.
// Q is pre-scaled by 1/sqrt(DH)=0.125 at staging. K tile staged async.
// ---------------------------------------------------------------------------
__global__ __launch_bounds__(128) void attn1_kernel(
    const unsigned short* __restrict__ qb, const unsigned short* __restrict__ kb,
    const unsigned short* __restrict__ vb, unsigned short* __restrict__ ctx,
    float* __restrict__ Mst, float* __restrict__ Rst) {
  constexpr int DP = 72;  // 64 + 8 pad (144B rows)
  __shared__ unsigned short Qs[64][DP];
  __shared__ unsigned short Ks[64][DP];
  __shared__ unsigned short Vt[64][DP];      // transposed: Vt[d][k]
  __shared__ unsigned short Ps[4][16][DP];   // per-wave P tile

  const int tid = threadIdx.x, lane = tid & 31, wid = tid >> 5;
  const int lrow = lane & 15, rbase = 8 * (lane >> 4);
  const int qt = blockIdx.x, h = blockIdx.y, b = blockIdx.z;
  const int q0 = qt * 64;

  {  // stage Q once, scaled by 0.125
    const int row = tid >> 1, half = tid & 1;
    const uint4* src = reinterpret_cast<const uint4*>(
        qb + ((size_t)(b * 1024 + q0 + row)) * 1024 + h * 64 + half * 32);
    unsigned short* dst = &Qs[row][half * 32];
#pragma unroll
    for (int q4 = 0; q4 < 4; ++q4) store8_scaled(dst + q4 * 8, src[q4], 0.125f);
  }

  float m_i[8], l_i[8];
#pragma unroll
  for (int r = 0; r < 8; ++r) { m_i[r] = -3.0e38f; l_i[r] = 0.0f; }
  v8f o[4];
#pragma unroll
  for (int j = 0; j < 4; ++j) o[j] = vzero();

  for (int kt = 0; kt < 16; ++kt) {
    __syncthreads();  // protect Ks/Vt (and Qs on first iter)
    {  // stage K tile: async global->LDS (pure bf16 copy)
      const int row = tid >> 1, half = tid & 1;
      const unsigned short* src =
          kb + ((size_t)(b * 1024 + kt * 64 + row)) * 1024 + h * 64 + half * 32;
      unsigned short* dst = &Ks[row][half * 32];
#pragma unroll
      for (int q4 = 0; q4 < 4; ++q4)
        async_ld_b128(dst + q4 * 8, src + q4 * 8);
    }
    {  // stage V transposed (VALU scatter)
      const int k = tid >> 1, half = tid & 1;
      const uint4* src = reinterpret_cast<const uint4*>(
          vb + ((size_t)(b * 1024 + kt * 64 + k)) * 1024 + h * 64 + half * 32);
#pragma unroll
      for (int q4 = 0; q4 < 4; ++q4) {
        uint4 v = src[q4];
        unsigned wv[4] = {v.x, v.y, v.z, v.w};
#pragma unroll
        for (int c = 0; c < 4; ++c) {
          const int d = half * 32 + q4 * 8 + c * 2;
          Vt[d][k]     = (unsigned short)(wv[c] & 0xffffu);
          Vt[d + 1][k] = (unsigned short)(wv[c] >> 16);
        }
      }
    }
    wait_async0();
    __syncthreads();

    // S = (Q*scale) K^T : 4 col subtiles x 2 K-steps
    v8f s[4];
#pragma unroll
    for (int j = 0; j < 4; ++j) {
      s[j] = vzero();
#pragma unroll
      for (int kk = 0; kk < 2; ++kk) {
        v16bf a  = frag_a(&Qs[wid * 16 + lrow][0], kk * 32, lane);
        v16bf bb = frag_b(&Ks[j * 16 + lrow][0], kk * 32, lane);
        s[j] = wmma_bf16(a, bb, s[j]);
      }
    }

    // online softmax
    float mnew[8], alpha[8];
#pragma unroll
    for (int r = 0; r < 8; ++r) {
      float t = fmaxf(fmaxf(s[0][r], s[1][r]), fmaxf(s[2][r], s[3][r]));
#pragma unroll
      for (int off = 1; off < 16; off <<= 1) t = fmaxf(t, __shfl_xor(t, off));
      mnew[r] = fmaxf(m_i[r], t);
      alpha[r] = __expf(m_i[r] - mnew[r]);
      m_i[r] = mnew[r];
    }
    float rs[8];
#pragma unroll
    for (int r = 0; r < 8; ++r) rs[r] = 0.0f;
#pragma unroll
    for (int j = 0; j < 4; ++j)
#pragma unroll
      for (int r = 0; r < 8; ++r) {
        float p = __expf(s[j][r] - mnew[r]);
        s[j][r] = p;
        rs[r] += p;
      }
#pragma unroll
    for (int r = 0; r < 8; ++r) {
      float t = rs[r];
#pragma unroll
      for (int off = 1; off < 16; off <<= 1) t += __shfl_xor(t, off);
      l_i[r] = l_i[r] * alpha[r] + t;
    }
#pragma unroll
    for (int j = 0; j < 4; ++j)
#pragma unroll
      for (int r = 0; r < 8; ++r) o[j][r] *= alpha[r];

    // P (C-layout) -> LDS -> A-fragment layout
#pragma unroll
    for (int j = 0; j < 4; ++j)
#pragma unroll
      for (int r = 0; r < 8; ++r)
        Ps[wid][rbase + r][j * 16 + lrow] = f2bf(s[j][r]);
    __syncthreads();

    // O += P V
#pragma unroll
    for (int j = 0; j < 4; ++j)
#pragma unroll
      for (int kk = 0; kk < 2; ++kk) {
        v16bf a  = frag_a(&Ps[wid][lrow][0], kk * 32, lane);
        v16bf bb = frag_b(&Vt[j * 16 + lrow][0], kk * 32, lane);
        o[j] = wmma_bf16(a, bb, o[j]);
      }
  }

  float rinv[8];
#pragma unroll
  for (int r = 0; r < 8; ++r) rinv[r] = 1.0f / l_i[r];
#pragma unroll
  for (int j = 0; j < 4; ++j)
#pragma unroll
    for (int r = 0; r < 8; ++r) {
      const int row = q0 + wid * 16 + rbase + r;
      ctx[((size_t)(b * 1024 + row)) * 1024 + h * 64 + j * 16 + lrow] =
          f2bf(o[j][r] * rinv[r]);
    }
  if (lrow == 0) {
#pragma unroll
    for (int r = 0; r < 8; ++r) {
      const int row = q0 + wid * 16 + rbase + r;
      const size_t idx = ((size_t)(b * 16 + h)) * 1024 + row;
      Mst[idx] = m_i[r];
      Rst[idx] = rinv[r];
    }
  }
}

// ---------------------------------------------------------------------------
// attn2: attn_weights = mean_h softmax(S). Recompute S via WMMA using saved
// stats; accumulate over heads in registers (no atomics, deterministic).
// Grid (kstrip=8, qtile=64, b=8); block 128 = 4 waves; wave owns 32 cols.
// K strip staged via async global->LDS.
// ---------------------------------------------------------------------------
__global__ __launch_bounds__(128) void attn2_kernel(
    const unsigned short* __restrict__ qb, const unsigned short* __restrict__ kb,
    const float* __restrict__ Mst, const float* __restrict__ Rst,
    float* __restrict__ attnw) {
  constexpr int DP = 72;
  __shared__ unsigned short Qs[16][DP];
  __shared__ unsigned short Ks[128][DP];
  const int tid = threadIdx.x, lane = tid & 31, wid = tid >> 5;
  const int lrow = lane & 15, rbase = 8 * (lane >> 4);
  const int ks = blockIdx.x, qt = blockIdx.y, b = blockIdx.z;
  const int q0 = qt * 16, n0 = ks * 128;

  v8f acc[2];
  acc[0] = vzero(); acc[1] = vzero();

  for (int h = 0; h < 16; ++h) {
    __syncthreads();
#pragma unroll
    for (int p = 0; p < 8; ++p) {  // stage K strip (128 rows) async
      const int idx = p * 128 + tid;
      const int row = idx >> 3, seg = idx & 7;
      async_ld_b128(&Ks[row][seg * 8],
                    kb + ((size_t)(b * 1024 + n0 + row)) * 1024 + h * 64 + seg * 8);
    }
    {  // stage Q (16 rows), scaled
      const int row = tid >> 3, seg = tid & 7;
      uint4 v = *reinterpret_cast<const uint4*>(
          qb + ((size_t)(b * 1024 + q0 + row)) * 1024 + h * 64 + seg * 8);
      store8_scaled(&Qs[row][seg * 8], v, 0.125f);
    }
    wait_async0();
    __syncthreads();

    float mrow[8], rrow[8];
#pragma unroll
    for (int r = 0; r < 8; ++r) {
      const size_t sidx = ((size_t)(b * 16 + h)) * 1024 + q0 + rbase + r;
      mrow[r] = Mst[sidx];
      rrow[r] = Rst[sidx];
    }
#pragma unroll
    for (int j = 0; j < 2; ++j) {
      v8f s = vzero();
#pragma unroll
      for (int kk = 0; kk < 2; ++kk) {
        v16bf a  = frag_a(&Qs[lrow][0], kk * 32, lane);
        v16bf bb = frag_b(&Ks[wid * 32 + j * 16 + lrow][0], kk * 32, lane);
        s = wmma_bf16(a, bb, s);
      }
#pragma unroll
      for (int r = 0; r < 8; ++r)
        acc[j][r] += __expf(s[r] - mrow[r]) * rrow[r] * 0.0625f;
    }
  }

#pragma unroll
  for (int j = 0; j < 2; ++j)
#pragma unroll
    for (int r = 0; r < 8; ++r) {
      const int row = q0 + rbase + r;
      const int col = n0 + wid * 32 + j * 16 + lrow;
      attnw[((size_t)(b * 1024 + row)) * 1024 + col] = acc[j][r];
    }
}

// ---------------------------------------------------------------------------
// In-place LayerNorm over E=1024 per row.
// ---------------------------------------------------------------------------
__global__ __launch_bounds__(256) void ln_kernel(float* __restrict__ x,
                                                 const float* __restrict__ g,
                                                 const float* __restrict__ bta) {
  const int row = blockIdx.x;
  const int t = threadIdx.x;
  const size_t base = (size_t)row * 1024;
  float v[4];
  float s = 0.0f, s2 = 0.0f;
#pragma unroll
  for (int i = 0; i < 4; ++i) {
    v[i] = x[base + t + i * 256];
    s += v[i];
    s2 += v[i] * v[i];
  }
#pragma unroll
  for (int off = 1; off < 32; off <<= 1) {
    s += __shfl_xor(s, off);
    s2 += __shfl_xor(s2, off);
  }
  __shared__ float red[2][8];
  const int wid = t >> 5, lane = t & 31;
  if (lane == 0) { red[0][wid] = s; red[1][wid] = s2; }
  __syncthreads();
  float ts = 0.0f, ts2 = 0.0f;
#pragma unroll
  for (int i = 0; i < 8; ++i) { ts += red[0][i]; ts2 += red[1][i]; }
  const float mu = ts * (1.0f / 1024.0f);
  const float var = ts2 * (1.0f / 1024.0f) - mu * mu;
  const float rstd = rsqrtf(var + 1e-5f);
#pragma unroll
  for (int i = 0; i < 4; ++i) {
    const int c = t + i * 256;
    x[base + c] = (v[i] - mu) * rstd * g[c] + bta[c];
  }
}

// ---------------------------------------------------------------------------
extern "C" void kernel_launch(void* const* d_in, const int* in_sizes, int n_in,
                              void* d_out, int out_size, void* d_ws, size_t ws_size,
                              hipStream_t stream) {
  (void)in_sizes; (void)n_in; (void)out_size; (void)ws_size;
  const float* query = (const float*)d_in[0];
  const float* key_t = (const float*)d_in[1];
  const float* value = (const float*)d_in[2];
  const float* ipw   = (const float*)d_in[3];  // [3E, E]
  const float* ipb   = (const float*)d_in[4];  // [3E]
  const float* opw   = (const float*)d_in[5];  // [E, E]
  const float* opb   = (const float*)d_in[6];  // [E]
  const float* gamma = (const float*)d_in[7];
  const float* beta  = (const float*)d_in[8];

  const size_t NE = (size_t)8 * 1024 * 1024;  // B*Lq*E elements
  char* w = (char*)d_ws;
  unsigned short* qbuf = (unsigned short*)w; w += NE * 2;
  unsigned short* kbuf = (unsigned short*)w; w += NE * 2;
  unsigned short* vbuf = (unsigned short*)w; w += NE * 2;
  unsigned short* ctx  = (unsigned short*)w; w += NE * 2;
  float* Mst = (float*)w; w += (size_t)8 * 16 * 1024 * 4;
  float* Rst = (float*)w; w += (size_t)8 * 16 * 1024 * 4;

  float* y     = (float*)d_out;       // pre-LN x, then y in place
  float* attnw = (float*)d_out + NE;  // mean-over-heads probs

  const dim3 gproj(1024 / 128, 8192 / 128);  // (N tiles, M tiles)
  gemm_wmma_kernel<false, false><<<gproj, 256, 0, stream>>>(
      query, ipw, ipb, nullptr, qbuf, 8192, 1024, 1024);
  gemm_wmma_kernel<false, false><<<gproj, 256, 0, stream>>>(
      key_t, ipw + 1024 * 1024, ipb + 1024, nullptr, kbuf, 8192, 1024, 1024);
  gemm_wmma_kernel<false, false><<<gproj, 256, 0, stream>>>(
      value, ipw + 2 * 1024 * 1024, ipb + 2048, nullptr, vbuf, 8192, 1024, 1024);

  attn1_kernel<<<dim3(16, 16, 8), 128, 0, stream>>>(qbuf, kbuf, vbuf, ctx, Mst, Rst);
  attn2_kernel<<<dim3(8, 64, 8), 128, 0, stream>>>(qbuf, kbuf, Mst, Rst, attnw);

  gemm_wmma_kernel<true, true><<<gproj, 256, 0, stream>>>(
      ctx, opw, opb, query, y, 8192, 1024, 1024);
  ln_kernel<<<8192, 256, 0, stream>>>(y, gamma, beta);
}
